// multihead_self_attention_2929167696378
// MI455X (gfx1250) — matrix operand order repaired and checked by reading, not compiled, not run on hardware
//
#include <hip/hip_runtime.h>

typedef unsigned short u16;
typedef unsigned int   u32;

typedef __attribute__((ext_vector_type(16))) __bf16 v16bf;
typedef __attribute__((ext_vector_type(8)))  float  v8f;

union Frag { v16bf v; u32   u[8]; };
union Acc  { v8f   v; float f[8]; };

// ---------------------------------------------------------------- helpers

// Round-to-nearest f32 -> bf16, result left in bits [31:16].
// One VALU op; consumers use (u16)(x >> 16) so the compiler emits d16_hi stores.
__device__ __forceinline__ u32 f2bf_hi(float x) {
  return __float_as_uint(x) + 0x8000u;
}

__device__ __forceinline__ u16 f2bf(float x) {
  return (u16)(f2bf_hi(x) >> 16);
}

// Pack two f32 -> {bf16(b) , bf16(a)} in one dword: 2 adds + v_perm_b32.
__device__ __forceinline__ u32 f2bf_pack(float a, float b) {
  return __builtin_amdgcn_perm(f2bf_hi(b), f2bf_hi(a), 0x07060302u);
}

__device__ __forceinline__ v8f wmma_bf16(const Frag& a, const Frag& b, v8f c) {
  return __builtin_amdgcn_wmma_f32_16x16x32_bf16(
      /*neg_a=*/false, a.v, /*neg_b=*/false, b.v,
      /*c_mod=*/(short)0, c, /*reuse_a=*/false, /*reuse_b=*/false);
}

// A-fragment: 16(M) x 32(K) bf16.  LDS row-major dword view, row pitch `pitch`,
// K-chunk dword base `dwb`.  Lanes 0-15: K 0..15; lanes 16-31: K 16..31.
__device__ __forceinline__ Frag load_fragA(const u32* lds, int row0, int dwb, int pitch) {
  int lane = threadIdx.x & 31;
  int hf   = lane >> 4;
  const u32* p = lds + (size_t)(row0 + (lane & 15)) * pitch + dwb + (hf << 3);
  Frag f;
  f.u[0] = p[0]; f.u[1] = p[1]; f.u[2] = p[2];  f.u[3] = p[3];
  f.u[4] = p[4]; f.u[5] = p[5]; f.u[6] = p[6]; f.u[7] = p[7];
  return f;
}

// B-fragment: 32(K) x 16(N) bf16.  LDS holds B as [N][K] rows (N-major):
// lane n reads K contiguously; lanes 0-15 K0..15, lanes 16-31 K16..31.
__device__ __forceinline__ Frag load_fragB(const u32* lds, int col0, int dwb, int pitch) {
  int lane = threadIdx.x & 31;
  int hf   = lane >> 4;
  const u32* p = lds + (size_t)(col0 + (lane & 15)) * pitch + dwb + (hf << 3);
  Frag f;
#pragma unroll
  for (int j = 0; j < 8; ++j) f.u[j] = p[j];
  return f;
}

// ---------------------------------------------------------------- fp32 -> bf16 pack

__global__ void cvt_f32_bf16(const float* __restrict__ in, u32* __restrict__ out, int n2) {
  int i = blockIdx.x * blockDim.x + threadIdx.x;
  if (i < n2) {
    float a = in[2 * i], b = in[2 * i + 1];
    out[i] = f2bf_pack(a, b);
  }
}

// ---------------------------------------------------------------- GEMM  C[m,n] = sum_k A[m,k]*B[n,k]
// A: [4096][1024] bf16 (dword view), B: [1024][1024] bf16 row-major (N-major tiles).
// Block tile 128x128, 8 waves (4 m x 2 n), wave tile 32x64.  LDS double-buffered.
// EPI 0: fp32 C[m][n].   EPI 1: bf16 into Vt[b][h][d][s] (V transpose for attention).

template <int EPI>
__global__ __launch_bounds__(256) void gemm_bf16(const u32* __restrict__ A,
                                                 const u32* __restrict__ Bm,
                                                 float* __restrict__ Cf,
                                                 u16* __restrict__ Cb) {
  const int PITCH = 20;                       // dwords; 80B rows: 16B-aligned + conflict-free
  __shared__ u32 la[2][128 * PITCH];
  __shared__ u32 lb[2][128 * PITCH];
  const int Kdw = 512;                        // K=1024 bf16 = 512 dwords
  int tid  = threadIdx.x;
  int lane = tid & 31, wid = tid >> 5;
  int wm = wid & 3, wn = wid >> 2;            // 4x2 wave grid, 32x64 per wave
  int nb = blockIdx.x & 7, mb = blockIdx.x >> 3;
  int m0 = mb * 128, n0 = nb * 128;

  // Per-thread copy slots: 128 rows x 4 quads -> each thread moves 2 quads per matrix.
  int r0 = tid >> 2, c0 = (tid & 3) << 2;

  Acc acc[2][4];
#pragma unroll
  for (int i = 0; i < 2; ++i)
#pragma unroll
    for (int j = 0; j < 4; ++j)
#pragma unroll
      for (int e = 0; e < 8; ++e) acc[i][j].f[e] = 0.f;

  auto load_tile = [&](int buf, int k0) {
    *(uint4*)&la[buf][(size_t)r0 * PITCH + c0] =
        *(const uint4*)&A[(size_t)(m0 + r0) * Kdw + k0 + c0];
    *(uint4*)&la[buf][(size_t)(r0 + 64) * PITCH + c0] =
        *(const uint4*)&A[(size_t)(m0 + r0 + 64) * Kdw + k0 + c0];
    *(uint4*)&lb[buf][(size_t)r0 * PITCH + c0] =
        *(const uint4*)&Bm[(size_t)(n0 + r0) * Kdw + k0 + c0];
    *(uint4*)&lb[buf][(size_t)(r0 + 64) * PITCH + c0] =
        *(const uint4*)&Bm[(size_t)(n0 + r0 + 64) * Kdw + k0 + c0];
  };

  load_tile(0, 0);
  __syncthreads();
  int p = 0;
  for (int k0 = 0; k0 < Kdw; k0 += 16) {      // BK = 32 bf16
    if (k0 + 16 < Kdw) load_tile(p ^ 1, k0 + 16);
    int kpf = (k0 + 32 < Kdw) ? k0 + 32 : k0; // clamped 2-step-ahead prefetch
    __builtin_prefetch(&A[(size_t)(m0 + r0) * Kdw + kpf + c0], 0, 3);
    __builtin_prefetch(&Bm[(size_t)(n0 + r0) * Kdw + kpf + c0], 0, 3);

    Frag fa0 = load_fragA(la[p], wm * 32, 0, PITCH);
    Frag fa1 = load_fragA(la[p], wm * 32 + 16, 0, PITCH);
#pragma unroll
    for (int ni = 0; ni < 4; ++ni) {
      Frag fb = load_fragB(lb[p], wn * 64 + ni * 16, 0, PITCH);
      acc[0][ni].v = wmma_bf16(fa0, fb, acc[0][ni].v);
      acc[1][ni].v = wmma_bf16(fa1, fb, acc[1][ni].v);
    }
    __syncthreads();
    p ^= 1;
  }

  int hf = lane >> 4, ln = lane & 15;
#pragma unroll
  for (int mi = 0; mi < 2; ++mi)
#pragma unroll
    for (int ni = 0; ni < 4; ++ni)
#pragma unroll
      for (int e = 0; e < 8; ++e) {
        int r = m0 + wm * 32 + mi * 16 + e + hf * 8;
        int c = n0 + wn * 64 + ni * 16 + ln;
        float val = acc[mi][ni].f[e];
        if (EPI == 0) {
          Cf[(size_t)r * 1024 + c] = val;
        } else {                               // V transpose: Vt[((b*16+h)*64+d)][s]
          int bb = r >> 11, s = r & 2047;
          int h = c >> 6, d = c & 63;
          Cb[((size_t)((bb * 16 + h) * 64 + d) << 11) + s] = f2bf(val);
        }
      }
}

// ---------------------------------------------------------------- fused RMSNorm + RoPE
// In: fp32 [4096][1024].  Out: bf16 [b][h][2048][64].  outscale folds 1/sqrt(DK) for Q.

__global__ __launch_bounds__(256) void rmsnorm_rope(const float* __restrict__ In,
                                                    const float* __restrict__ gain,
                                                    const int* __restrict__ positions,
                                                    const int* __restrict__ thetap,
                                                    u16* __restrict__ Out, float outscale) {
  __shared__ float red[8];
  int row = blockIdx.x;                       // b*S + s
  int b = row >> 11, s = row & 2047;
  int tid = threadIdx.x;
  const float* x = In + (size_t)row * 1024;
  float4 xv = *(const float4*)&x[tid * 4];
  float ssq = xv.x * xv.x + xv.y * xv.y + xv.z * xv.z + xv.w * xv.w;
#pragma unroll
  for (int m = 16; m >= 1; m >>= 1) ssq += __shfl_xor(ssq, m, 32);
  if ((tid & 31) == 0) red[tid >> 5] = ssq;
  __syncthreads();
  float tot = 0.f;
#pragma unroll
  for (int i = 0; i < 8; ++i) tot += red[i];
  float inv = rsqrtf(tot * (1.0f / 1024.0f) + 1e-5f);

  float pos = (float)positions[s];
  float lt  = __logf((float)thetap[0]) * (1.0f / 32.0f);
  int e0 = tid * 4;
  float4 gv = *(const float4*)&gain[e0];
  float ys[4] = { xv.x * inv * gv.x, xv.y * inv * gv.y,
                  xv.z * inv * gv.z, xv.w * inv * gv.w };
  int h = e0 >> 6, dk = e0 & 63;
  size_t obase = (((size_t)(b * 16 + h) * 2048 + s) * 64 + dk) >> 1;  // dword index
  u32* O32 = (u32*)Out;
#pragma unroll
  for (int pp = 0; pp < 2; ++pp) {            // two interleaved (even,odd) pairs
    int i = (dk + 2 * pp) >> 1;               // rotary index 0..31 within head
    float ang = pos * __expf(-(float)i * lt);
    float sn, cs;
    __sincosf(ang, &sn, &cs);
    float re = (ys[2 * pp] * cs - ys[2 * pp + 1] * sn) * outscale;
    float ro = (ys[2 * pp + 1] * cs + ys[2 * pp] * sn) * outscale;
    O32[obase + pp] = f2bf_pack(re, ro);
  }
}

// ---------------------------------------------------------------- flash attention
// grid: (b*h)*32 q-tiles of 64. 4 waves; wave w owns q rows [q0+16w, q0+16w+16).
// Q,K bf16 [bh][2048][64]; Vt bf16 [bh][64][2048]; AO bf16 [b][s][1024].

__global__ __launch_bounds__(128) void attn_fwd(const u16* __restrict__ Qh,
                                                const u16* __restrict__ Kh,
                                                const u16* __restrict__ Vt,
                                                u16* __restrict__ AO) {
  const int LP = 36;                          // 144B rows: 16B-aligned + conflict-free
  __shared__ u32 lq[64 * LP];
  __shared__ u32 lk[64 * LP];
  __shared__ u32 lv[64 * LP];
  __shared__ u32 lp[4 * 16 * LP];             // per-wave P tile
  int tid = threadIdx.x;
  int lane = tid & 31, wid = tid >> 5;
  int hf = lane >> 4, ln = lane & 15;
  int qt = blockIdx.x & 31;
  int bh = blockIdx.x >> 5;
  int b = bh >> 4, h = bh & 15;
  int q0 = qt * 64;
  const u32* Q32 = (const u32*)(Qh + (size_t)bh * 2048 * 64);
  const u32* K32 = (const u32*)(Kh + (size_t)bh * 2048 * 64);
  const u32* V32 = (const u32*)(Vt + (size_t)bh * 64 * 2048);

  for (int t = tid; t < 64 * 8; t += 128) {   // Q tile, once
    int r = t >> 3, c4 = (t & 7) << 2;
    *(uint4*)&lq[r * LP + c4] = *(const uint4*)&Q32[(size_t)(q0 + r) * 32 + c4];
  }

  Acc oacc[4];
  float m_run[8], l_run[8];
#pragma unroll
  for (int e = 0; e < 8; ++e) { m_run[e] = -3.0e38f; l_run[e] = 0.f; }
#pragma unroll
  for (int n = 0; n < 4; ++n)
#pragma unroll
    for (int e = 0; e < 8; ++e) oacc[n].f[e] = 0.f;

  int kend = q0 + 64;
  for (int k0 = 0; k0 < kend; k0 += 64) {
    __syncthreads();
    {
      int r = tid >> 1, c4 = (tid & 1) << 4;  // prefetch next K tile (clamped)
      int kn = (k0 + 64 < kend) ? k0 + 64 : k0;
      __builtin_prefetch(&K32[(size_t)(kn + r) * 32 + c4], 0, 3);
    }
    for (int t = tid; t < 64 * 8; t += 128) {
      int r = t >> 3, c4 = (t & 7) << 2;
      *(uint4*)&lk[r * LP + c4] = *(const uint4*)&K32[(size_t)(k0 + r) * 32 + c4];
      *(uint4*)&lv[r * LP + c4] = *(const uint4*)&V32[(size_t)r * 1024 + (k0 >> 1) + c4];
    }
    __syncthreads();

    Acc sacc[4];
#pragma unroll
    for (int n = 0; n < 4; ++n)
#pragma unroll
      for (int e = 0; e < 8; ++e) sacc[n].f[e] = 0.f;
#pragma unroll
    for (int c = 0; c < 2; ++c) {             // DK=64 -> 2 K-chunks of 32
      Frag fa = load_fragA(lq, wid * 16, c * 16, LP);
#pragma unroll
      for (int n = 0; n < 4; ++n) {
        Frag fb = load_fragB(lk, n * 16, c * 16, LP);
        sacc[n].v = wmma_bf16(fa, fb, sacc[n].v);
      }
    }

    if (k0 + 63 > q0 + wid * 16) {            // causal mask (diagonal tiles only)
#pragma unroll
      for (int n = 0; n < 4; ++n)
#pragma unroll
        for (int e = 0; e < 8; ++e) {
          int kg = k0 + n * 16 + ln;
          int qg = q0 + wid * 16 + e + hf * 8;
          if (kg > qg) sacc[n].f[e] = -3.0e38f;
        }
    }

    float rmax[8], rsum[8];
#pragma unroll
    for (int e = 0; e < 8; ++e) {             // row max across 16-lane half
      float v = fmaxf(fmaxf(sacc[0].f[e], sacc[1].f[e]),
                      fmaxf(sacc[2].f[e], sacc[3].f[e]));
#pragma unroll
      for (int m = 8; m >= 1; m >>= 1) v = fmaxf(v, __shfl_xor(v, m, 32));
      rmax[e] = v;
    }
#pragma unroll
    for (int e = 0; e < 8; ++e) {             // online-softmax rescale
      float mnew = fmaxf(m_run[e], rmax[e]);
      float sc = __expf(m_run[e] - mnew);
      l_run[e] *= sc;
#pragma unroll
      for (int n = 0; n < 4; ++n) oacc[n].f[e] *= sc;
      m_run[e] = mnew;
    }
#pragma unroll
    for (int e = 0; e < 8; ++e) rsum[e] = 0.f;
#pragma unroll
    for (int n = 0; n < 4; ++n)
#pragma unroll
      for (int e = 0; e < 8; ++e) {
        float p = __expf(sacc[n].f[e] - m_run[e]);
        sacc[n].f[e] = p;
        rsum[e] += p;
      }
#pragma unroll
    for (int e = 0; e < 8; ++e) {
      float v = rsum[e];
#pragma unroll
      for (int m = 8; m >= 1; m >>= 1) v += __shfl_xor(v, m, 32);
      l_run[e] += v;
    }

    // P (C-layout) -> per-wave LDS tile (row-major) so it can re-enter as A-fragment.
    // 1 VALU + ds_store_b16_d16_hi per element.
    u16* lps = (u16*)&lp[wid * 16 * LP];
#pragma unroll
    for (int n = 0; n < 4; ++n)
#pragma unroll
      for (int e = 0; e < 8; ++e)
        lps[(e + hf * 8) * (LP * 2) + n * 16 + ln] = (u16)(f2bf_hi(sacc[n].f[e]) >> 16);

#pragma unroll
    for (int c = 0; c < 2; ++c) {             // O += P @ V  (K = 64 keys -> 2 chunks)
      Frag fa = load_fragA(lp + wid * 16 * LP, 0, c * 16, LP);
#pragma unroll
      for (int n = 0; n < 4; ++n) {
        Frag fb = load_fragB(lv, n * 16, c * 16, LP);
        oacc[n].v = wmma_bf16(fa, fb, oacc[n].v);
      }
    }
  }

#pragma unroll
  for (int e = 0; e < 8; ++e) l_run[e] = 1.0f / l_run[e];
#pragma unroll
  for (int n = 0; n < 4; ++n)
#pragma unroll
    for (int e = 0; e < 8; ++e) {
      int qg = q0 + wid * 16 + e + hf * 8;
      size_t idx = ((size_t)(b * 2048 + qg)) * 1024 + h * 64 + n * 16 + ln;
      AO[idx] = (u16)(f2bf_hi(oacc[n].f[e] * l_run[e]) >> 16);
    }
}

// ---------------------------------------------------------------- launch

extern "C" void kernel_launch(void* const* d_in, const int* in_sizes, int n_in,
                              void* d_out, int out_size, void* d_ws, size_t ws_size,
                              hipStream_t stream) {
  (void)in_sizes; (void)n_in; (void)out_size; (void)ws_size;
  const float* x    = (const float*)d_in[0];
  const float* Wq   = (const float*)d_in[1];
  const float* Wk   = (const float*)d_in[2];
  const float* Wv   = (const float*)d_in[3];
  const float* Wo   = (const float*)d_in[4];
  const float* gain = (const float*)d_in[5];
  const int* positions = (const int*)d_in[6];
  const int* theta     = (const int*)d_in[7];

  char* ws = (char*)d_ws;
  size_t off = 0;
  auto alloc = [&](size_t bytes) {
    void* p = ws + off;
    off += (bytes + 255) & ~(size_t)255;
    return p;
  };
  u32* xb   = (u32*)alloc((size_t)4096 * 1024 * 2);   // x bf16
  u32* wqb  = (u32*)alloc((size_t)1024 * 1024 * 2);
  u32* wkb  = (u32*)alloc((size_t)1024 * 1024 * 2);
  u32* wvb  = (u32*)alloc((size_t)1024 * 1024 * 2);
  u32* wob  = (u32*)alloc((size_t)1024 * 1024 * 2);
  float* Qf = (float*)alloc((size_t)4096 * 1024 * 4); // pre-norm Q
  float* Kf = (float*)alloc((size_t)4096 * 1024 * 4); // pre-norm K
  u16* Qhb  = (u16*)alloc((size_t)4096 * 1024 * 2);   // [b][h][s][dk]
  u16* Khb  = (u16*)alloc((size_t)4096 * 1024 * 2);
  u16* Vtb  = (u16*)alloc((size_t)4096 * 1024 * 2);   // [b][h][dk][s]
  u16* AOb  = (u16*)alloc((size_t)4096 * 1024 * 2);   // attn out [b][s][D]

  cvt_f32_bf16<<<(4096 * 1024 / 2) / 256, 256, 0, stream>>>(x,  xb,  4096 * 1024 / 2);
  cvt_f32_bf16<<<(1024 * 1024 / 2) / 256, 256, 0, stream>>>(Wq, wqb, 1024 * 1024 / 2);
  cvt_f32_bf16<<<(1024 * 1024 / 2) / 256, 256, 0, stream>>>(Wk, wkb, 1024 * 1024 / 2);
  cvt_f32_bf16<<<(1024 * 1024 / 2) / 256, 256, 0, stream>>>(Wv, wvb, 1024 * 1024 / 2);
  cvt_f32_bf16<<<(1024 * 1024 / 2) / 256, 256, 0, stream>>>(Wo, wob, 1024 * 1024 / 2);

  gemm_bf16<0><<<256, 256, 0, stream>>>(xb, wqb, Qf, nullptr);
  gemm_bf16<0><<<256, 256, 0, stream>>>(xb, wkb, Kf, nullptr);
  gemm_bf16<1><<<256, 256, 0, stream>>>(xb, wvb, nullptr, Vtb);

  rmsnorm_rope<<<4096, 256, 0, stream>>>(Qf, gain, positions, theta, Qhb, 0.125f); // 1/sqrt(64)
  rmsnorm_rope<<<4096, 256, 0, stream>>>(Kf, gain, positions, theta, Khb, 1.0f);

  attn_fwd<<<1024, 128, 0, stream>>>(Qhb, Khb, Vtb, AOb);

  gemm_bf16<0><<<256, 256, 0, stream>>>((const u32*)AOb, wob, (float*)d_out, nullptr);
}
